// TritonByteEncoder_27212912788374
// MI455X (gfx1250) — matrix operand order repaired and checked by reading, not compiled, not run on hardware
//
#include <hip/hip_runtime.h>

// ---------------- problem constants ----------------
#define B_SZ 16
#define T_SZ 2048
#define D_SZ 1024
#define F_SZ 2048          // 2*D
#define PI_TRUNC 3.14159f
#define LN_EPS_F 1e-5f

typedef __attribute__((ext_vector_type(16))) __bf16          v16bf;
typedef __attribute__((ext_vector_type(8)))  float           v8f;
typedef __attribute__((ext_vector_type(16))) unsigned short  v16u;
typedef __attribute__((ext_vector_type(8)))  unsigned short  v8u;
typedef __attribute__((ext_vector_type(4)))  int             v4i;

// ---------------- CDNA5 async global->LDS (guarded; sync fallback) ----------------
#if defined(__has_builtin)
#if __has_builtin(__builtin_amdgcn_global_load_async_to_lds_b128) && \
    __has_builtin(__builtin_amdgcn_s_wait_asynccnt)
#define HAVE_ASYNC_LDS 1
#endif
#endif
#ifndef HAVE_ASYNC_LDS
#define HAVE_ASYNC_LDS 0
#endif

using gv4i_p = __attribute__((address_space(1))) v4i*;   // global src
using lv4i_p = __attribute__((address_space(3))) v4i*;   // LDS dst

// per-lane copy of 32 bytes (16 bf16) from global to LDS
__device__ __forceinline__ void stage_row(const unsigned short* g, unsigned short* l) {
#if HAVE_ASYNC_LDS
  __builtin_amdgcn_global_load_async_to_lds_b128((gv4i_p)(g),     (lv4i_p)(l),     0, 0);
  __builtin_amdgcn_global_load_async_to_lds_b128((gv4i_p)(g + 8), (lv4i_p)(l + 8), 0, 0);
#else
  *(v8u*)l       = *(const v8u*)g;
  *(v8u*)(l + 8) = *(const v8u*)(g + 8);
#endif
}
__device__ __forceinline__ void async_wait() {
#if HAVE_ASYNC_LDS
  __builtin_amdgcn_s_wait_asynccnt(0);
#endif
}

// ---------------- workspace layout (bytes) ----------------
static constexpr size_t OFF_W1T  = 0;                                   // F x D bf16 (W1^T)
static constexpr size_t OFF_W2T  = OFF_W1T  + (size_t)F_SZ * D_SZ * 2;  // D x F bf16 (W2^T)
static constexpr size_t OFF_MBF  = OFF_W2T  + (size_t)D_SZ * F_SZ * 2;  // T x D bf16
static constexpr size_t OFF_G    = OFF_MBF  + (size_t)T_SZ * D_SZ * 2;  // T x F f32
static constexpr size_t OFF_GM   = OFF_G    + (size_t)T_SZ * F_SZ * 4;  // T f32
static constexpr size_t OFF_VG   = OFF_GM   + (size_t)T_SZ * 4;
static constexpr size_t OFF_COV  = OFF_VG   + (size_t)T_SZ * 4;
static constexpr size_t OFF_SABS = OFF_COV  + (size_t)T_SZ * 4;         // 16 f32
static constexpr size_t OFF_B1ST = OFF_SABS + 64;                       // 2 f32 (mean, var)

// ---------------- helpers ----------------
__device__ __forceinline__ unsigned short f2bf(float x) {
  unsigned u = __float_as_uint(x);
  u += 0x7FFFu + ((u >> 16) & 1u);          // round-to-nearest-even
  return (unsigned short)(u >> 16);
}
__device__ __forceinline__ float gelu_exact(float x) {
  return 0.5f * x * (1.0f + erff(x * 0.70710678118654752f));
}
__device__ __forceinline__ float block_reduce(float v, float* red) {
  const int tid = threadIdx.x;
  red[tid] = v; __syncthreads();
  for (int off = 128; off; off >>= 1) {
    if (tid < off) red[tid] += red[tid + off];
    __syncthreads();
  }
  float r = red[0]; __syncthreads();
  return r;
}

// ---------------- stage 0: |S_b| ----------------
__global__ __launch_bounds__(256) void k_sabs(const int* __restrict__ ids,
                                              float* __restrict__ sabs) {
  __shared__ float red[256];
  const int b = blockIdx.x;
  int s = 0;
  for (int i = threadIdx.x; i < T_SZ; i += 256) s += ids[(size_t)b * T_SZ + i];
  float tot = block_reduce((float)s, red);
  if (threadIdx.x == 0) sabs[b] = fabsf(tot / 127.5f - (float)T_SZ);
}

// ---------------- stage 0b: b1 mean / var ----------------
__global__ __launch_bounds__(256) void k_b1stats(const float* __restrict__ b1,
                                                 float* __restrict__ b1st) {
  __shared__ float red[256];
  float s = 0.f;
  for (int i = threadIdx.x; i < F_SZ; i += 256) s += b1[i];
  float mean = block_reduce(s, red) / (float)F_SZ;
  float sv = 0.f;
  for (int i = threadIdx.x; i < F_SZ; i += 256) { float d = b1[i] - mean; sv += d * d; }
  float var = block_reduce(sv, red) / (float)F_SZ;
  if (threadIdx.x == 0) { b1st[0] = mean; b1st[1] = var; }
}

// ---------------- stage 0c: f32 -> bf16 with transpose (LDS tiled) ----------------
// in: R x C f32 ; outT: C x R bf16
__global__ __launch_bounds__(256) void k_transpose_bf(const float* __restrict__ in,
                                                      unsigned short* __restrict__ outT,
                                                      int R, int C) {
  __shared__ unsigned short tile[32][33];
  const int c0 = blockIdx.x * 32, r0 = blockIdx.y * 32;
  const int tx = threadIdx.x & 31, ty = threadIdx.x >> 5;   // 32 x 8
#pragma unroll
  for (int i = 0; i < 32; i += 8)
    tile[ty + i][tx] = f2bf(in[(size_t)(r0 + ty + i) * C + c0 + tx]);
  __syncthreads();
#pragma unroll
  for (int i = 0; i < 32; i += 8)
    outT[(size_t)(c0 + ty + i) * R + r0 + tx] = tile[tx][ty + i];
}

// ---------------- stage 0d: M[t,d] = |unit magnitude| in bf16 ----------------
__global__ __launch_bounds__(256) void k_makeM(unsigned short* __restrict__ Mbf) {
  const size_t i = (size_t)blockIdx.x * 256 + threadIdx.x;  // T*D elements
  const int t = (int)(i >> 10);          // / D_SZ
  const int d = (int)(i & (D_SZ - 1));
  const int freq = (t * d) & (T_SZ - 1); // (t*d) % T, T power of 2
  const float phase = (2.0f * PI_TRUNC / (float)T_SZ) * (float)freq;
  const float c = cosf(phase), s = sinf(phase);
  Mbf[i] = f2bf(sqrtf(c * c + s * s));
}

// ---------------- fragment loaders (bf16, ISA 16-bit layouts) ----------------
// A (16x32): m=lane&15; lanes<16 hold K{0..7,16..23}, lanes>=16 K{8..15,24..31}
__device__ __forceinline__ v16u load_a_frag(const unsigned short* As, int m, int kHalf) {
  v8u r1 = *(const v8u*)&As[m * 40 + 8 * kHalf];
  v8u r2 = *(const v8u*)&As[m * 40 + 16 + 8 * kHalf];
  return __builtin_shufflevector(r1, r2, 0,1,2,3,4,5,6,7,8,9,10,11,12,13,14,15);
}
// B (32x16), LDS stored [n][k]: lanes<16 hold K{0..7,16..23}, lanes>=16 K{8..15,24..31}
__device__ __forceinline__ v16u load_b_frag(const unsigned short* Bs, int n, int kHalf) {
  v8u r1 = *(const v8u*)&Bs[n * 40 + 8 * kHalf];
  v8u r2 = *(const v8u*)&Bs[n * 40 + 16 + 8 * kHalf];
  return __builtin_shufflevector(r1, r2, 0,1,2,3,4,5,6,7,8,9,10,11,12,13,14,15);
}

// ---------------- stage 1: G = M @ W1  (bf16 WMMA, ping-pong LDS, async stage) ----------------
__global__ __launch_bounds__(256) void k_gemm1(const unsigned short* __restrict__ Mbf,
                                               const unsigned short* __restrict__ W1T,
                                               float* __restrict__ G) {
  __shared__ __align__(16) unsigned short A_s[2][128 * 40];
  __shared__ __align__(16) unsigned short B_s[2][128 * 40];
  const int tid = threadIdx.x;
  const int lane = tid & 31, wave = tid >> 5;
  const int waveM = wave >> 2, waveN = wave & 3;   // 2 x 4 waves
  const int mLane = lane & 15, kHalf = lane >> 4;
  const int t0 = blockIdx.y * 128;
  const int n0 = blockIdx.x * 128;

  const int sr = tid >> 1, sc0 = (tid & 1) * 16;   // staging: 128 rows x 32 k

  v8f acc[4][2] = {};

  auto stage = [&](int kk, int buf) {
    stage_row(&Mbf[(size_t)(t0 + sr) * D_SZ + kk + sc0], &A_s[buf][sr * 40 + sc0]);
    stage_row(&W1T[(size_t)(n0 + sr) * D_SZ + kk + sc0], &B_s[buf][sr * 40 + sc0]);
  };

  stage(0, 0);
  int p = 0;
  for (int k0 = 0; k0 < D_SZ; k0 += 32) {
    async_wait();
    __syncthreads();
    if (k0 + 32 < D_SZ) stage(k0 + 32, p ^ 1);   // overlap DMA with WMMA below

    v16u afr[4], bfr[2];
#pragma unroll
    for (int mi = 0; mi < 4; ++mi)
      afr[mi] = load_a_frag(&A_s[p][0], waveM * 64 + mi * 16 + mLane, kHalf);
#pragma unroll
    for (int ni = 0; ni < 2; ++ni)
      bfr[ni] = load_b_frag(&B_s[p][0], waveN * 32 + ni * 16 + mLane, kHalf);

#pragma unroll
    for (int mi = 0; mi < 4; ++mi)
#pragma unroll
      for (int ni = 0; ni < 2; ++ni)
        acc[mi][ni] = __builtin_amdgcn_wmma_f32_16x16x32_bf16(
            false, __builtin_bit_cast(v16bf, afr[mi]),
            false, __builtin_bit_cast(v16bf, bfr[ni]),
            (short)0, acc[mi][ni], false, false);
    p ^= 1;
  }

#pragma unroll
  for (int mi = 0; mi < 4; ++mi)
#pragma unroll
    for (int ni = 0; ni < 2; ++ni) {
      const int colg = n0 + waveN * 32 + ni * 16 + mLane;
#pragma unroll
      for (int v = 0; v < 8; ++v) {
        const int rowg = t0 + waveM * 64 + mi * 16 + v + 8 * kHalf;
        G[(size_t)rowg * F_SZ + colg] = acc[mi][ni][v];
      }
    }
}

// ---------------- stage 2: per-row stats of G ----------------
__global__ __launch_bounds__(256) void k_gstats(const float* __restrict__ G,
                                                const float* __restrict__ b1,
                                                const float* __restrict__ b1st,
                                                float* __restrict__ Gm,
                                                float* __restrict__ Vg,
                                                float* __restrict__ Cov) {
  __shared__ float red[256];
  const int t = blockIdx.x;
  const float b1m = b1st[0];
  float s = 0.f;
  for (int f = threadIdx.x; f < F_SZ; f += 256) s += G[(size_t)t * F_SZ + f];
  const float gm = block_reduce(s, red) / (float)F_SZ;
  float sv = 0.f, sc = 0.f;
  for (int f = threadIdx.x; f < F_SZ; f += 256) {
    const float d = G[(size_t)t * F_SZ + f] - gm;
    sv += d * d;
    sc += d * (b1[f] - b1m);
  }
  const float vg  = block_reduce(sv, red) / (float)F_SZ;
  const float cov = block_reduce(sc, red) / (float)F_SZ;
  if (threadIdx.x == 0) { Gm[t] = gm; Vg[t] = vg; Cov[t] = cov; }
}

// ---------------- stage 3: out = gelu(LN(s*G + b1)) @ W2 + b2 ----------------
__global__ __launch_bounds__(256) void k_gemm2(const float* __restrict__ G,
                                               const unsigned short* __restrict__ W2T,
                                               const float* __restrict__ b1,
                                               const float* __restrict__ gam,
                                               const float* __restrict__ bet,
                                               const float* __restrict__ b2,
                                               const float* __restrict__ Gm,
                                               const float* __restrict__ Vg,
                                               const float* __restrict__ Cov,
                                               const float* __restrict__ sabs,
                                               const float* __restrict__ b1st,
                                               float* __restrict__ out) {
  __shared__ __align__(16) unsigned short A_s[2][128 * 40];
  __shared__ __align__(16) unsigned short B_s[2][128 * 40];
  const int tid = threadIdx.x;
  const int lane = tid & 31, wave = tid >> 5;
  const int waveM = wave >> 2, waveN = wave & 3;
  const int mLane = lane & 15, kHalf = lane >> 4;
  const int r0 = blockIdx.y * 128;   // rows of (B*T)
  const int n0 = blockIdx.x * 128;   // cols of D

  // per-thread staging row (fixed across k loop)
  const int ar = tid >> 1, ac0 = (tid & 1) * 16;
  const int rowg = r0 + ar;
  const int b = rowg >> 11;          // / T_SZ
  const int t = rowg & (T_SZ - 1);
  const float s    = sabs[b];
  const float gm   = Gm[t];
  const float b1m  = b1st[0];
  const float var  = s * s * Vg[t] + 2.f * s * Cov[t] + b1st[1];
  const float rinv = rsqrtf(var + LN_EPS_F);

  v8f acc[4][2] = {};

  // fused LayerNorm + exact GELU -> bf16, packed 2 x ds_store_b128
  auto stageA = [&](int kk, int buf) {
    const float4* gp = (const float4*)&G[(size_t)t * F_SZ + kk + ac0];
    float4 q0 = gp[0], q1 = gp[1], q2 = gp[2], q3 = gp[3];
    const float gv[16] = { q0.x,q0.y,q0.z,q0.w, q1.x,q1.y,q1.z,q1.w,
                           q2.x,q2.y,q2.z,q2.w, q3.x,q3.y,q3.z,q3.w };
    if (kk + 32 < F_SZ)
      __builtin_prefetch(&G[(size_t)t * F_SZ + kk + 32 + ac0], 0, 1);
    v8u t0v, t1v;
#pragma unroll
    for (int i = 0; i < 16; ++i) {
      const int f = kk + ac0 + i;
      const float dev = s * (gv[i] - gm) + (b1[f] - b1m);
      const float hn  = dev * rinv * gam[f] + bet[f];
      const unsigned short h = f2bf(gelu_exact(hn));
      if (i < 8) t0v[i] = h; else t1v[i - 8] = h;
    }
    *(v8u*)&A_s[buf][ar * 40 + ac0]     = t0v;
    *(v8u*)&A_s[buf][ar * 40 + ac0 + 8] = t1v;
  };
  auto stageB = [&](int kk, int buf) {
    stage_row(&W2T[(size_t)(n0 + ar) * F_SZ + kk + ac0], &B_s[buf][ar * 40 + ac0]);
  };

  stageB(0, 0);
  stageA(0, 0);
  int p = 0;
  for (int k0 = 0; k0 < F_SZ; k0 += 32) {
    async_wait();
    __syncthreads();
    if (k0 + 32 < F_SZ) { stageB(k0 + 32, p ^ 1); stageA(k0 + 32, p ^ 1); }

    v16u afr[4], bfr[2];
#pragma unroll
    for (int mi = 0; mi < 4; ++mi)
      afr[mi] = load_a_frag(&A_s[p][0], waveM * 64 + mi * 16 + mLane, kHalf);
#pragma unroll
    for (int ni = 0; ni < 2; ++ni)
      bfr[ni] = load_b_frag(&B_s[p][0], waveN * 32 + ni * 16 + mLane, kHalf);

#pragma unroll
    for (int mi = 0; mi < 4; ++mi)
#pragma unroll
      for (int ni = 0; ni < 2; ++ni)
        acc[mi][ni] = __builtin_amdgcn_wmma_f32_16x16x32_bf16(
            false, __builtin_bit_cast(v16bf, afr[mi]),
            false, __builtin_bit_cast(v16bf, bfr[ni]),
            (short)0, acc[mi][ni], false, false);
    p ^= 1;
  }

#pragma unroll
  for (int mi = 0; mi < 4; ++mi)
#pragma unroll
    for (int ni = 0; ni < 2; ++ni) {
      const int colg = n0 + waveN * 32 + ni * 16 + mLane;
      const float bias = b2[colg];
#pragma unroll
      for (int v = 0; v < 8; ++v) {
        const int rg = r0 + waveM * 64 + mi * 16 + v + 8 * kHalf;
        out[(size_t)rg * D_SZ + colg] = acc[mi][ni][v] + bias;
      }
    }
}

// ---------------- host-side launch ----------------
extern "C" void kernel_launch(void* const* d_in, const int* in_sizes, int n_in,
                              void* d_out, int out_size, void* d_ws, size_t ws_size,
                              hipStream_t stream) {
  (void)in_sizes; (void)n_in; (void)out_size; (void)ws_size;
  const int*   byte_ids = (const int*)  d_in[0];
  const float* W1  = (const float*)d_in[1];
  const float* b1  = (const float*)d_in[2];
  const float* gam = (const float*)d_in[3];
  const float* bet = (const float*)d_in[4];
  const float* W2  = (const float*)d_in[5];
  const float* b2  = (const float*)d_in[6];

  char* ws = (char*)d_ws;
  unsigned short* W1T = (unsigned short*)(ws + OFF_W1T);
  unsigned short* W2T = (unsigned short*)(ws + OFF_W2T);
  unsigned short* Mbf = (unsigned short*)(ws + OFF_MBF);
  float* G    = (float*)(ws + OFF_G);
  float* Gm   = (float*)(ws + OFF_GM);
  float* Vg   = (float*)(ws + OFF_VG);
  float* Cov  = (float*)(ws + OFF_COV);
  float* sabs = (float*)(ws + OFF_SABS);
  float* b1st = (float*)(ws + OFF_B1ST);

  k_sabs   <<<B_SZ, 256, 0, stream>>>(byte_ids, sabs);
  k_b1stats<<<1,    256, 0, stream>>>(b1, b1st);
  // W1 (DxF) -> W1T (FxD) bf16 ; W2 (FxD) -> W2T (DxF) bf16
  k_transpose_bf<<<dim3(F_SZ / 32, D_SZ / 32), 256, 0, stream>>>(W1, W1T, D_SZ, F_SZ);
  k_transpose_bf<<<dim3(D_SZ / 32, F_SZ / 32), 256, 0, stream>>>(W2, W2T, F_SZ, D_SZ);
  k_makeM  <<<(T_SZ * D_SZ) / 256, 256, 0, stream>>>(Mbf);
  k_gemm1  <<<dim3(F_SZ / 128, T_SZ / 128), 256, 0, stream>>>(Mbf, W1T, G);
  k_gstats <<<T_SZ, 256, 0, stream>>>(G, b1, b1st, Gm, Vg, Cov);
  k_gemm2  <<<dim3(D_SZ / 128, (B_SZ * T_SZ) / 128), 256, 0, stream>>>(
      G, W2T, b1, gam, bet, b2, Gm, Vg, Cov, sabs, b1st, (float*)d_out);
}
